// BiLSTM_CRF_22007412424867
// MI455X (gfx1250) — matrix operand order repaired and checked by reading, not compiled, not run on hardware
//
#include <hip/hip_runtime.h>
#include <hip/hip_bf16.h>
#include <math.h>

typedef __attribute__((ext_vector_type(16))) _Float16 v16h;
typedef __attribute__((ext_vector_type(8)))  float    v8f;

#define LSEQ   512
#define BATCH  64
#define EMBD   256
#define HIDF   512      // 2*H (concat)
#define HHALF  256      // H
#define G4     1024     // 4*H
#define NTAG   32
#define MTOT   (LSEQ*BATCH)   // 32768

// ---------------- WMMA helpers (CDNA5 gfx1250, wave32) ----------------

__device__ __forceinline__ v8f wmma_f16(v16h a, v16h b, v8f c) {
  // (neg_a, A, neg_b, B, c_mod, C, reuse_a, reuse_b)
  return __builtin_amdgcn_wmma_f32_16x16x32_f16(false, a, false, b, (short)0, c,
                                                false, false);
}

// A-matrix fragment 16x32 f16, source row-major [*, stride].
// lanes 0..15: M=lane, K=0..15; lanes 16..31: M=lane-16, K=16..31; K pairs contiguous -> b32 loads.
__device__ __forceinline__ v16h load_a16(const _Float16* src, int stride,
                                         int mbase, int kbase, int lane) {
  int m    = mbase + (lane & 15);
  int koff = (lane >> 4) * 16;
  union { v16h v; unsigned u[8]; } r;
#pragma unroll
  for (int vv = 0; vv < 8; ++vv) {
    int k = kbase + koff + vv * 2;
    r.u[vv] = *(const unsigned*)(src + (size_t)m * stride + k);
  }
  return r.v;
}

// B-matrix fragment 32x16 f16 from row-major W[N,K]: B[k][n] = W[n][k].
// lanes 0..15: n=lane, K=0..15; lanes 16..31: n=lane-16, K=16..31.
__device__ __forceinline__ v16h load_b16(const _Float16* W, int stride,
                                         int nbase, int kbase, int lane) {
  int n  = nbase + (lane & 15);
  int kb = kbase + (lane >> 4) * 16;
  union { v16h v; unsigned u[8]; } r;
#pragma unroll
  for (int vv = 0; vv < 8; ++vv)
    r.u[vv] = *(const unsigned*)(W + (size_t)n * stride + kb + 2 * vv);
  return r.v;
}

__device__ __forceinline__ float sigmoidf(float x) {
  return 1.0f / (1.0f + __expf(-x));
}

// ---------------- Kernel 1: weight conversion to f16 + combined bias ----------------

__global__ void prep_weights(const float* Wih_f, const float* Whh_f,
                             const float* bih_f, const float* bhh_f,
                             const float* Wih_b, const float* Whh_b,
                             const float* bih_b, const float* bhh_b,
                             const float* Wout,
                             _Float16* Wih16, _Float16* Whh16,
                             _Float16* Wout16, float* bias2) {
  int idx = blockIdx.x * blockDim.x + threadIdx.x;
  const int NW = G4 * EMBD;                // 262144 per dir (Wih, Whh same size)
  if (idx < NW) {
    Wih16[idx]      = (_Float16)Wih_f[idx];
    Wih16[NW + idx] = (_Float16)Wih_b[idx];
    Whh16[idx]      = (_Float16)Whh_f[idx];
    Whh16[NW + idx] = (_Float16)Whh_b[idx];
  }
  if (idx < NTAG * HIDF) Wout16[idx] = (_Float16)Wout[idx];
  if (idx < G4) {
    bias2[idx]      = bih_f[idx] + bhh_f[idx];
    bias2[G4 + idx] = bih_b[idx] + bhh_b[idx];
  }
}

// ---------------- Kernel 2: embedding gather -> f16 ----------------

__global__ void embed_gather(const int* __restrict__ seqs,
                             const float* __restrict__ table,
                             _Float16* __restrict__ emb16) {
  int idx = blockIdx.x * blockDim.x + threadIdx.x;   // MTOT*EMBD threads
  int row = idx >> 8;            // l*B + b
  int e   = idx & 255;
  int tok = seqs[row];
  emb16[idx] = (_Float16)table[(size_t)tok * EMBD + e];
}

// ---------------- Kernel 3: x_proj GEMM (both dirs), +bias ----------------
// M=32768, K=256, N=2048 (fwd 0..1023, bwd 1024..2047). One wave: 1 Mtile x 8 Ntiles.

__global__ void xproj_gemm(const _Float16* __restrict__ emb16,
                           const _Float16* __restrict__ Wih16,
                           const float* __restrict__ bias2,
                           float* __restrict__ xproj) {
  int lane = threadIdx.x & 31;
  int wid  = threadIdx.x >> 5;
  int w    = blockIdx.x * (blockDim.x >> 5) + wid;   // 0..32767
  int mtile  = w >> 4;                               // 0..2047
  int nbase0 = (w & 15) * 128;                       // 0..1920, stays within a dir
  int d      = nbase0 >> 10;

  v8f acc[8];
#pragma unroll
  for (int j = 0; j < 8; ++j)
#pragma unroll
    for (int r = 0; r < 8; ++r) acc[j][r] = 0.0f;

  const _Float16* W = Wih16 + (size_t)d * G4 * EMBD;
#pragma unroll
  for (int k = 0; k < 8; ++k) {
    v16h a = load_a16(emb16, EMBD, mtile * 16, k * 32, lane);
#pragma unroll
    for (int j = 0; j < 8; ++j) {
      int g = (nbase0 & 1023) + j * 16;
      v16h b = load_b16(W, EMBD, g, k * 32, lane);
      acc[j] = wmma_f16(a, b, acc[j]);
    }
  }
#pragma unroll
  for (int j = 0; j < 8; ++j) {
    int ncol = nbase0 + j * 16 + (lane & 15);
    float bv = bias2[ncol];
#pragma unroll
    for (int r = 0; r < 8; ++r) {
      int m = mtile * 16 + r + ((lane >> 4) << 3);
      xproj[(size_t)m * 2048 + ncol] = acc[j][r] + bv;
    }
  }
}

// ---------------- Kernel 4: LSTM recurrence (1 block per direction) ----------------
// 32 waves: wave owns hh range (wid&15)*16..+16 for ALL 4 gates, and 2 M-tiles
// (wid>>4). Cell math fully in registers; h state in LDS (f16), c in registers.

__global__ __launch_bounds__(1024, 1)
void lstm_rec(const float* __restrict__ xproj,
              const _Float16* __restrict__ Whh16,
              const int* __restrict__ masks,
              _Float16* __restrict__ out16) {
  int d    = blockIdx.x;
  int lane = threadIdx.x & 31;
  int wid  = threadIdx.x >> 5;
  int hh16  = (wid & 15) * 16;
  int mhalf = wid >> 4;                 // M-tiles 2*mhalf, 2*mhalf+1
  int hh    = hh16 + (lane & 15);

  __shared__ _Float16 hbuf[BATCH * HHALF];   // 32 KB
  for (int i = threadIdx.x; i < BATCH * HHALF; i += blockDim.x)
    hbuf[i] = (_Float16)0.0f;

  float cst[2][8];
#pragma unroll
  for (int mt = 0; mt < 2; ++mt)
#pragma unroll
    for (int r = 0; r < 8; ++r) cst[mt][r] = 0.0f;

  const _Float16* W = Whh16 + (size_t)d * G4 * HHALF;
  __syncthreads();

  for (int t = 0; t < LSEQ; ++t) {
    int l = d ? (LSEQ - 1 - t) : t;
    v8f acc[2][4];
    // accumulator init = x_proj (already includes b_ih + b_hh)
#pragma unroll
    for (int mt = 0; mt < 2; ++mt)
#pragma unroll
      for (int q = 0; q < 4; ++q) {
        int ncol = q * 256 + hh;
#pragma unroll
        for (int r = 0; r < 8; ++r) {
          int b = (mhalf * 2 + mt) * 16 + r + ((lane >> 4) << 3);
          acc[mt][q][r] =
              xproj[((size_t)(l * BATCH + b)) * 2048 + d * 1024 + ncol];
        }
      }
    // gates += h_{t-1} @ Whh^T   (K = 256 -> 8 WMMA steps)
#pragma unroll
    for (int k = 0; k < 8; ++k) {
      v16h a0 = load_a16(hbuf, HHALF, (mhalf * 2 + 0) * 16, k * 32, lane);
      v16h a1 = load_a16(hbuf, HHALF, (mhalf * 2 + 1) * 16, k * 32, lane);
#pragma unroll
      for (int q = 0; q < 4; ++q) {
        v16h b = load_b16(W, HHALF, q * 256 + hh16, k * 32, lane);
        acc[0][q] = wmma_f16(a0, b, acc[0][q]);
        acc[1][q] = wmma_f16(a1, b, acc[1][q]);
      }
    }
    // LSTM cell — each lane owns i,f,g,o of the same (b, hh)
    float hsel[2][8];
#pragma unroll
    for (int mt = 0; mt < 2; ++mt)
#pragma unroll
      for (int r = 0; r < 8; ++r) {
        int b = (mhalf * 2 + mt) * 16 + r + ((lane >> 4) << 3);
        float i_ = sigmoidf(acc[mt][0][r]);
        float f_ = sigmoidf(acc[mt][1][r]);
        float g_ = tanhf(acc[mt][2][r]);
        float o_ = sigmoidf(acc[mt][3][r]);
        float cn = f_ * cst[mt][r] + i_ * g_;
        float hn = o_ * tanhf(cn);
        int  m   = masks[l * BATCH + b];
        float hp = (float)hbuf[b * HHALF + hh];
        hsel[mt][r] = m ? hn : hp;
        cst[mt][r]  = m ? cn : cst[mt][r];
        out16[((size_t)(l * BATCH + b)) * HIDF + d * HHALF + hh] =
            (_Float16)(m ? hn : 0.0f);
      }
    __syncthreads();          // everyone done reading h_{t-1}
#pragma unroll
    for (int mt = 0; mt < 2; ++mt)
#pragma unroll
      for (int r = 0; r < 8; ++r) {
        int b = (mhalf * 2 + mt) * 16 + r + ((lane >> 4) << 3);
        hbuf[b * HHALF + hh] = (_Float16)hsel[mt][r];
      }
    __syncthreads();          // h_t visible to all
  }
}

// ---------------- Kernel 5: emissions GEMM (32768,512)x(32,512)^T ----------------

__global__ void emis_gemm(const _Float16* __restrict__ lstm16,
                          const _Float16* __restrict__ Wout16,
                          const float* __restrict__ b_out,
                          float* __restrict__ emis) {
  int lane  = threadIdx.x & 31;
  int wid   = threadIdx.x >> 5;
  int mtile = blockIdx.x * (blockDim.x >> 5) + wid;   // 0..2047
  v8f acc[2];
#pragma unroll
  for (int nt = 0; nt < 2; ++nt)
#pragma unroll
    for (int r = 0; r < 8; ++r) acc[nt][r] = 0.0f;
#pragma unroll
  for (int k = 0; k < 16; ++k) {
    v16h a = load_a16(lstm16, HIDF, mtile * 16, k * 32, lane);
#pragma unroll
    for (int nt = 0; nt < 2; ++nt) {
      v16h b = load_b16(Wout16, HIDF, nt * 16, k * 32, lane);
      acc[nt] = wmma_f16(a, b, acc[nt]);
    }
  }
#pragma unroll
  for (int nt = 0; nt < 2; ++nt)
#pragma unroll
    for (int r = 0; r < 8; ++r) {
      int m = mtile * 16 + r + ((lane >> 4) << 3);
      int n = nt * 16 + (lane & 15);
      emis[(size_t)m * NTAG + n] = acc[nt][r] + b_out[n];
    }
}

// ---------------- Kernel 6: CRF numerator + partition (1 wave per batch elt) ----

__global__ void crf_kernel(const float* __restrict__ emis,
                           const int* __restrict__ tags,
                           const int* __restrict__ masks,
                           const float* __restrict__ start_trans,
                           const float* __restrict__ end_trans,
                           const float* __restrict__ trans,
                           float* __restrict__ llh) {
  int b = blockIdx.x;
  int j = threadIdx.x;     // 0..31
  __shared__ float tr[NTAG * NTAG];
  __shared__ float sc[NTAG];
  __shared__ float red[NTAG];
  __shared__ float red2[NTAG];
  __shared__ int   redi[NTAG];
  for (int i = j; i < NTAG * NTAG; i += 32) tr[i] = trans[i];

  float score = start_trans[j] + emis[(size_t)b * NTAG + j];
  sc[j] = score;
  __syncthreads();

  for (int l = 1; l < LSEQ; ++l) {
    int   m = masks[l * BATCH + b];
    float e = emis[((size_t)l * BATCH + b) * NTAG + j];
    float mx = -1e30f;
    for (int i = 0; i < NTAG; ++i) mx = fmaxf(mx, sc[i] + tr[i * NTAG + j]);
    float s = 0.0f;
    for (int i = 0; i < NTAG; ++i) s += __expf(sc[i] + tr[i * NTAG + j] - mx);
    float nxt = mx + __logf(s) + e;
    score = m ? nxt : score;
    __syncthreads();
    sc[j] = score;
    __syncthreads();
  }

  red[j] = score + end_trans[j];
  // numerator partial sums (strided over l) + mask-length partial
  float part = 0.0f;
  int   msum = 0;
  for (int l = j; l < LSEQ; l += 32) msum += masks[l * BATCH + b];
  for (int l = 1 + j; l < LSEQ; l += 32) {
    int tl = tags[l * BATCH + b];
    int tp = tags[(l - 1) * BATCH + b];
    float mf = (float)masks[l * BATCH + b];
    part += (emis[((size_t)l * BATCH + b) * NTAG + tl] + tr[tp * NTAG + tl]) * mf;
  }
  red2[j] = part;
  redi[j] = msum;
  __syncthreads();

  if (j == 0) {
    float mx = red[0];
    for (int i = 1; i < NTAG; ++i) mx = fmaxf(mx, red[i]);
    float s = 0.0f;
    for (int i = 0; i < NTAG; ++i) s += __expf(red[i] - mx);
    float log_z = mx + __logf(s);
    float num = 0.0f;
    int length = 0;
    for (int i = 0; i < NTAG; ++i) { num += red2[i]; length += redi[i]; }
    int t0 = tags[b];
    int tl = tags[(length - 1) * BATCH + b];
    num += start_trans[t0] + emis[(size_t)b * NTAG + t0] + end_trans[tl];
    llh[b] = num - log_z;
  }
}

// ---------------- Kernel 7: final reduction -> -mean(llh) ----------------

__global__ void finalize(const float* __restrict__ llh, float* __restrict__ out) {
  __shared__ float red[BATCH];
  int t = threadIdx.x;
  red[t] = llh[t];
  __syncthreads();
  if (t == 0) {
    float s = 0.0f;
    for (int i = 0; i < BATCH; ++i) s += red[i];
    out[0] = -(s / (float)BATCH);
  }
}

// ---------------- Host launcher ----------------

extern "C" void kernel_launch(void* const* d_in, const int* in_sizes, int n_in,
                              void* d_out, int out_size, void* d_ws, size_t ws_size,
                              hipStream_t stream) {
  (void)in_sizes; (void)n_in; (void)out_size; (void)ws_size;
  const int*   seqs        = (const int*)d_in[0];
  const int*   tags        = (const int*)d_in[1];
  const int*   masks       = (const int*)d_in[2];
  const float* embed_table = (const float*)d_in[3];
  const float* W_ih_f      = (const float*)d_in[4];
  const float* W_hh_f      = (const float*)d_in[5];
  const float* b_ih_f      = (const float*)d_in[6];
  const float* b_hh_f      = (const float*)d_in[7];
  const float* W_ih_b      = (const float*)d_in[8];
  const float* W_hh_b      = (const float*)d_in[9];
  const float* b_ih_b      = (const float*)d_in[10];
  const float* b_hh_b      = (const float*)d_in[11];
  const float* W_out       = (const float*)d_in[12];
  const float* b_out       = (const float*)d_in[13];
  const float* start_trans = (const float*)d_in[14];
  const float* end_trans   = (const float*)d_in[15];
  const float* trans       = (const float*)d_in[16];

  char* p = (char*)d_ws;
  auto alloc = [&](size_t bytes) {
    char* r = p;
    p += (bytes + 255) & ~(size_t)255;
    return r;
  };
  _Float16* emb16  = (_Float16*)alloc((size_t)MTOT * EMBD * 2);      // 16.8 MB
  _Float16* Wih16  = (_Float16*)alloc((size_t)2 * G4 * EMBD * 2);    //  1.0 MB
  _Float16* Whh16  = (_Float16*)alloc((size_t)2 * G4 * HHALF * 2);   //  1.0 MB
  _Float16* Wout16 = (_Float16*)alloc((size_t)NTAG * HIDF * 2);      // 32 KB
  float*    bias2  = (float*)alloc((size_t)2 * G4 * 4);              //  8 KB
  float*    xproj  = (float*)alloc((size_t)MTOT * 2048 * 4);         // 268 MB
  _Float16* lstm16 = (_Float16*)alloc((size_t)MTOT * HIDF * 2);      // 33.5 MB
  float*    emis   = (float*)alloc((size_t)MTOT * NTAG * 4);         //  4.2 MB
  float*    llh    = (float*)alloc((size_t)BATCH * 4);

  prep_weights<<<(G4 * EMBD + 255) / 256, 256, 0, stream>>>(
      W_ih_f, W_hh_f, b_ih_f, b_hh_f, W_ih_b, W_hh_b, b_ih_b, b_hh_b, W_out,
      Wih16, Whh16, Wout16, bias2);

  embed_gather<<<(MTOT * EMBD) / 256, 256, 0, stream>>>(seqs, embed_table, emb16);

  // 32768 waves = 1 wave per (Mtile, Ngroup-of-8); 8 waves/block
  xproj_gemm<<<4096, 256, 0, stream>>>(emb16, Wih16, bias2, xproj);

  // one workgroup per direction, 32 waves each
  lstm_rec<<<2, 1024, 0, stream>>>(xproj, Whh16, masks, lstm16);

  // 2048 M-tiles, 8 waves/block
  emis_gemm<<<256, 256, 0, stream>>>(lstm16, Wout16, b_out, emis);

  crf_kernel<<<BATCH, 32, 0, stream>>>(emis, tags, masks, start_trans, end_trans,
                                       trans, llh);

  finalize<<<1, BATCH, 0, stream>>>(llh, (float*)d_out);
}
